// TransformerBlock_88235808129460
// MI455X (gfx1250) — compile-verified, matrix operand order checked by reading, not run on hardware
//
#include <hip/hip_runtime.h>
#include <math.h>

// ---------------- problem constants ----------------
#define DD    2048          // embedding
#define HH    16            // heads
#define HDD   128           // head dim
#define SS    2048          // seq len
#define BB    2             // batch
#define FFD   5632          // swiglu hidden
#define NTOK  (BB*SS)       // 4096 tokens
#define TD    (3*DD)        // 6144 (qkv out)

typedef __bf16 bf16_t;
typedef __attribute__((ext_vector_type(16))) __bf16 bf16x16;
typedef __attribute__((ext_vector_type(8)))  float  floatx8;

union BV { bf16x16 v; uint4 u[2]; };

__device__ __forceinline__ bf16x16 load_op(const bf16_t* p) {
    // A/B operand packing for v_wmma_*_16x16x32 (16-bit):
    // lane&15 = row/col index, lane>>4 selects K halves {0..7,16..23} vs {8..15,24..31}.
    // Caller passes p already offset by (lane>>4)*8 elements; we grab 2x16B chunks.
    BV r;
    r.u[0] = *(const uint4*)(p);
    r.u[1] = *(const uint4*)(p + 16);
    return r.v;
}

__device__ __forceinline__ floatx8 zero8() {
    floatx8 z;
#pragma unroll
    for (int i = 0; i < 8; ++i) z[i] = 0.0f;
    return z;
}

__device__ __forceinline__ float red_max16(float v) {
#pragma unroll
    for (int m = 1; m < 16; m <<= 1) v = fmaxf(v, __shfl_xor(v, m, 32));
    return v;
}
__device__ __forceinline__ float red_sum16(float v) {
#pragma unroll
    for (int m = 1; m < 16; m <<= 1) v += __shfl_xor(v, m, 32);
    return v;
}

// ---------------- elementwise helpers ----------------
__global__ __launch_bounds__(256) void f32_to_bf16_kernel(
    const float* __restrict__ src, bf16_t* __restrict__ dst, int n) {
    int i = blockIdx.x * 256 + threadIdx.x;
    if (i < n) dst[i] = (bf16_t)src[i];
}

__global__ __launch_bounds__(256) void add_kernel(
    const float* __restrict__ a, const float* __restrict__ b,
    float* __restrict__ out, int n) {
    int i = blockIdx.x * 256 + threadIdx.x;
    if (i < n) out[i] = a[i] + b[i];
}

__global__ __launch_bounds__(256) void swiglu_kernel(
    const float* __restrict__ g1, const float* __restrict__ g2,
    bf16_t* __restrict__ out, int n) {
    int i = blockIdx.x * 256 + threadIdx.x;
    if (i < n) {
        float a = g1[i];
        float s = a / (1.0f + __expf(-a));   // silu
        out[i] = (bf16_t)(s * g2[i]);
    }
}

// RMSNorm row kernel: out_bf16 = g * x * rsqrt(mean(x^2)+eps)
__global__ __launch_bounds__(256) void rmsnorm_bf16_kernel(
    const float* __restrict__ x, const float* __restrict__ g,
    bf16_t* __restrict__ out, int width) {
    const int row = blockIdx.x;
    const int tid = threadIdx.x;
    const float* xr = x + (size_t)row * width;
    float ss = 0.0f;
    for (int i = tid; i < width; i += 256) { float v = xr[i]; ss += v * v; }
    __shared__ float red[256];
    red[tid] = ss;
    __syncthreads();
    for (int off = 128; off > 0; off >>= 1) {
        if (tid < off) red[tid] += red[tid + off];
        __syncthreads();
    }
    const float inv = rsqrtf(red[0] / (float)width + 1e-6f);
    bf16_t* orow = out + (size_t)row * width;
    for (int i = tid; i < width; i += 256)
        orow[i] = (bf16_t)(g[i] * (xr[i] * inv));
}

// Split qkv (fp32 [tok,6144]) -> q/k row-major bf16 [b,h,s,hd] (+pe), v transposed [b,h,hd,s]
__global__ __launch_bounds__(256) void qkv_pack_kernel(
    const float* __restrict__ qkv, bf16_t* __restrict__ qb,
    bf16_t* __restrict__ kb, bf16_t* __restrict__ vT) {
    const int idx = blockIdx.x * 256 + threadIdx.x;
    if (idx >= NTOK * DD) return;
    const int token = idx >> 11;          // /2048
    const int m     = idx & 2047;
    const int h = m >> 7, hd = m & 127;
    const int b = token >> 11, s = token & 2047;
    // positional encoding row for pos = S-1 = 2047
    const float pair = (float)(m & ~1);
    const float dv   = __expf(pair * (-9.2103403719761836f / 2048.0f)); // -ln(10000)/D
    const float ang  = 2047.0f * dv;
    const float pe   = (m & 1) ? cosf(ang) : sinf(ang);
    const float* rp = qkv + (size_t)token * TD;
    const size_t bh = (size_t)(b * HH + h);
    qb[(bh * SS + s) * HDD + hd] = (bf16_t)(rp[m] + pe);
    kb[(bh * SS + s) * HDD + hd] = (bf16_t)(rp[DD + m] + pe);
    vT[(bh * HDD + hd) * SS + s] = (bf16_t)(rp[2 * DD + m]);
}

// ---------------- NT GEMM: C[N,M] = A[N,K] * B[M,K]^T + bias, bf16 in / f32 out ----------------
__global__ __launch_bounds__(128) void gemm_nt_bf16(
    const bf16_t* __restrict__ A, const bf16_t* __restrict__ Bw,
    const float* __restrict__ bias, float* __restrict__ C,
    int N, int M, int K) {
    const int lane = threadIdx.x & 31;
    const int wave = threadIdx.x >> 5;                 // 4 waves: 2x2 of 64x64 tiles
    const int n0 = blockIdx.y * 128 + (wave >> 1) * 64;
    const int m0 = blockIdx.x * 128 + (wave & 1) * 64;
    const int l  = lane & 15;
    const int hi = lane >> 4;

    floatx8 acc[4][4];
#pragma unroll
    for (int i = 0; i < 4; ++i)
#pragma unroll
        for (int j = 0; j < 4; ++j) acc[i][j] = zero8();

    for (int k0 = 0; k0 < K; k0 += 32) {
        bf16x16 av[4], bv[4];
#pragma unroll
        for (int i = 0; i < 4; ++i)
            av[i] = load_op(A + (size_t)(n0 + i * 16 + l) * K + k0 + hi * 8);
#pragma unroll
        for (int j = 0; j < 4; ++j)
            bv[j] = load_op(Bw + (size_t)(m0 + j * 16 + l) * K + k0 + hi * 8);
#pragma unroll
        for (int i = 0; i < 4; ++i)
#pragma unroll
            for (int j = 0; j < 4; ++j)
                acc[i][j] = __builtin_amdgcn_wmma_f32_16x16x32_bf16(
                    false, av[i], false, bv[j], (short)0, acc[i][j], false, false);
    }

    // C layout: lanes0-15 row = r, lanes16-31 row = r+8; col = lane&15
#pragma unroll
    for (int i = 0; i < 4; ++i) {
#pragma unroll
        for (int j = 0; j < 4; ++j) {
            const int col = m0 + j * 16 + l;
            const float bvl = bias ? bias[col] : 0.0f;
#pragma unroll
            for (int r = 0; r < 8; ++r) {
                const int row = n0 + i * 16 + r + 8 * hi;
                C[(size_t)row * M + col] = acc[i][j][r] + bvl;
            }
        }
    }
}

// ---------------- causal flash attention: one wave per 16-query tile per (b,h) ----------------
__global__ __launch_bounds__(32) void attention_kernel(
    const bf16_t* __restrict__ Qb, const bf16_t* __restrict__ Kb,
    const bf16_t* __restrict__ VTb, bf16_t* __restrict__ Ob) {
    const int lane = threadIdx.x;
    const int l = lane & 15, hi = lane >> 4;
    const int qt = blockIdx.x & (SS / 16 - 1);     // 128 q-tiles per (b,h)
    const int bh = blockIdx.x >> 7;
    const int b = bh >> 4, h = bh & 15;
    const bf16_t* Q  = Qb  + (size_t)bh * SS * HDD;
    const bf16_t* Kp = Kb  + (size_t)bh * SS * HDD;
    const bf16_t* VT = VTb + (size_t)bh * HDD * SS;
    const int q0 = qt * 16;

    bf16x16 qa[4];
#pragma unroll
    for (int c = 0; c < 4; ++c)
        qa[c] = load_op(Q + (size_t)(q0 + l) * HDD + c * 32 + hi * 8);

    floatx8 acc[8];
    float mrow[8], lrow[8];
#pragma unroll
    for (int j = 0; j < 8; ++j) acc[j] = zero8();
#pragma unroll
    for (int r = 0; r < 8; ++r) { mrow[r] = -1e30f; lrow[r] = 0.0f; }

    __shared__ __align__(16) bf16_t pbuf[16 * 32];
    const float scale = 0.088388347648318447f;     // 1/sqrt(128)

    const int ktiles = (q0 + 15) / 32 + 1;         // cover all keys <= q0+15
    for (int kt = 0; kt < ktiles; ++kt) {
        const int k0 = kt * 32;
        // scores 16x32 as two 16x16 tiles, K-chained over HD=128
        floatx8 s0 = zero8(), s1 = zero8();
#pragma unroll
        for (int c = 0; c < 4; ++c) {
            bf16x16 kv = load_op(Kp + (size_t)(k0 + l) * HDD + c * 32 + hi * 8);
            s0 = __builtin_amdgcn_wmma_f32_16x16x32_bf16(false, qa[c], false, kv, (short)0, s0, false, false);
        }
#pragma unroll
        for (int c = 0; c < 4; ++c) {
            bf16x16 kv = load_op(Kp + (size_t)(k0 + 16 + l) * HDD + c * 32 + hi * 8);
            s1 = __builtin_amdgcn_wmma_f32_16x16x32_bf16(false, qa[c], false, kv, (short)0, s1, false, false);
        }
        const bool need_mask = (k0 + 31 > q0);
        float alpha[8];
#pragma unroll
        for (int r = 0; r < 8; ++r) {
            float a0 = s0[r] * scale;
            float a1 = s1[r] * scale;
            if (need_mask) {
                const int row = q0 + r + 8 * hi;
                if (k0 + l      > row) a0 = -1e9f;
                if (k0 + 16 + l > row) a1 = -1e9f;
            }
            const float mx = red_max16(fmaxf(a0, a1));
            const float nm = fmaxf(mrow[r], mx);
            const float al = __expf(mrow[r] - nm);
            const float p0 = __expf(a0 - nm);
            const float p1 = __expf(a1 - nm);
            lrow[r] = lrow[r] * al + red_sum16(p0 + p1);
            mrow[r] = nm;
            alpha[r] = al;
            // stage P (C layout) into LDS row-major [16][32]
            const int prow = r + 8 * hi;
            pbuf[prow * 32 + l]      = (bf16_t)p0;
            pbuf[prow * 32 + 16 + l] = (bf16_t)p1;
        }
        // reload P in A-operand layout (DS ops same-wave are in order)
        bf16x16 pa = load_op(&pbuf[l * 32 + hi * 8]);
#pragma unroll
        for (int j = 0; j < 8; ++j) {
            bf16x16 vv = load_op(VT + (size_t)(j * 16 + l) * SS + k0 + hi * 8);
            floatx8 a = acc[j];
#pragma unroll
            for (int r = 0; r < 8; ++r) a[r] *= alpha[r];
            acc[j] = __builtin_amdgcn_wmma_f32_16x16x32_bf16(false, pa, false, vv, (short)0, a, false, false);
        }
    }
    // write [b,s,D] with D index = h*HD + d
#pragma unroll
    for (int j = 0; j < 8; ++j) {
#pragma unroll
        for (int r = 0; r < 8; ++r) {
            const int tok = q0 + r + 8 * hi;
            const int d   = j * 16 + l;
            Ob[((size_t)b * SS + tok) * DD + h * HDD + d] = (bf16_t)(acc[j][r] / lrow[r]);
        }
    }
}

// ---------------- orchestration ----------------
extern "C" void kernel_launch(void* const* d_in, const int* in_sizes, int n_in,
                              void* d_out, int out_size, void* d_ws, size_t ws_size,
                              hipStream_t stream) {
    (void)in_sizes; (void)n_in; (void)out_size; (void)ws_size;
    const float* x      = (const float*)d_in[0];
    // d_in[1] = mask (causal) -- handled analytically
    const float* qkv_w  = (const float*)d_in[2];
    const float* qkv_b  = (const float*)d_in[3];
    const float* fc_w   = (const float*)d_in[4];
    const float* fc_b   = (const float*)d_in[5];
    const float* w1     = (const float*)d_in[6];
    const float* b1     = (const float*)d_in[7];
    const float* w2     = (const float*)d_in[8];
    const float* b2     = (const float*)d_in[9];
    const float* w3     = (const float*)d_in[10];
    const float* b3     = (const float*)d_in[11];
    const float* g_attn = (const float*)d_in[12];
    const float* g_ff   = (const float*)d_in[13];
    float* out = (float*)d_out;

    char* ws = (char*)d_ws;
    size_t off = 0;
    auto alloc = [&](size_t bytes) -> char* {
        char* p = ws + off;
        off += (bytes + 255) & ~(size_t)255;
        return p;
    };
    bf16_t* qkvw_bf = (bf16_t*)alloc((size_t)TD * DD * 2);
    bf16_t* fcw_bf  = (bf16_t*)alloc((size_t)DD * DD * 2);
    bf16_t* w1_bf   = (bf16_t*)alloc((size_t)FFD * DD * 2);
    bf16_t* w2_bf   = (bf16_t*)alloc((size_t)FFD * DD * 2);
    bf16_t* w3_bf   = (bf16_t*)alloc((size_t)DD * FFD * 2);
    bf16_t* hin_bf  = (bf16_t*)alloc((size_t)NTOK * DD * 2);   // reused as fin_bf
    float*  qkv_f   = (float*)alloc((size_t)NTOK * TD * 4);    // reused as g1_f
    bf16_t* q_bf    = (bf16_t*)alloc((size_t)NTOK * DD * 2);   // gate_bf overlays q/k/vT
    bf16_t* k_bf    = (bf16_t*)alloc((size_t)NTOK * DD * 2);
    bf16_t* vT_bf   = (bf16_t*)alloc((size_t)NTOK * DD * 2);
    bf16_t* attn_bf = (bf16_t*)alloc((size_t)NTOK * DD * 2);
    float*  fc_f    = (float*)alloc((size_t)NTOK * DD * 4);    // reused as ff_f
    float*  h_f     = (float*)alloc((size_t)NTOK * DD * 4);
    float*  g2_f    = (float*)alloc((size_t)NTOK * FFD * 4);
    float*  g1_f    = qkv_f;
    bf16_t* fin_bf  = hin_bf;
    bf16_t* gate_bf = q_bf;
    float*  ff_f    = fc_f;

    auto cvt = [&](const float* s, bf16_t* d, int n) {
        f32_to_bf16_kernel<<<(n + 255) / 256, 256, 0, stream>>>(s, d, n);
    };
    cvt(qkv_w, qkvw_bf, TD * DD);
    cvt(fc_w,  fcw_bf,  DD * DD);
    cvt(w1,    w1_bf,   FFD * DD);
    cvt(w2,    w2_bf,   FFD * DD);
    cvt(w3,    w3_bf,   DD * FFD);

    // attention sublayer
    rmsnorm_bf16_kernel<<<NTOK, 256, 0, stream>>>(x, g_attn, hin_bf, DD);
    gemm_nt_bf16<<<dim3(TD / 128, NTOK / 128), 128, 0, stream>>>(hin_bf, qkvw_bf, qkv_b, qkv_f, NTOK, TD, DD);
    qkv_pack_kernel<<<(NTOK * DD + 255) / 256, 256, 0, stream>>>(qkv_f, q_bf, k_bf, vT_bf);
    attention_kernel<<<BB * HH * (SS / 16), 32, 0, stream>>>(q_bf, k_bf, vT_bf, attn_bf);
    gemm_nt_bf16<<<dim3(DD / 128, NTOK / 128), 128, 0, stream>>>(attn_bf, fcw_bf, fc_b, fc_f, NTOK, DD, DD);
    add_kernel<<<(NTOK * DD + 255) / 256, 256, 0, stream>>>(x, fc_f, h_f, NTOK * DD);

    // SwiGLU FFN sublayer
    rmsnorm_bf16_kernel<<<NTOK, 256, 0, stream>>>(h_f, g_ff, fin_bf, DD);
    gemm_nt_bf16<<<dim3(FFD / 128, NTOK / 128), 128, 0, stream>>>(fin_bf, w1_bf, b1, g1_f, NTOK, FFD, DD);
    gemm_nt_bf16<<<dim3(FFD / 128, NTOK / 128), 128, 0, stream>>>(fin_bf, w2_bf, b2, g2_f, NTOK, FFD, DD);
    swiglu_kernel<<<(NTOK * FFD + 255) / 256, 256, 0, stream>>>(g1_f, g2_f, gate_bf, NTOK * FFD);
    gemm_nt_bf16<<<dim3(DD / 128, NTOK / 128), 128, 0, stream>>>(gate_bf, w3_bf, b3, ff_f, NTOK, DD, FFD);
    add_kernel<<<(NTOK * DD + 255) / 256, 256, 0, stream>>>(h_f, ff_f, out, NTOK * DD);
}
